// SelfAttention_67980742361364
// MI455X (gfx1250) — matrix operand order repaired and checked by reading, not compiled, not run on hardware
//
#include <hip/hip_runtime.h>
#include <hip/hip_bf16.h>

// ---------------------------------------------------------------------------
// Shapes (fixed by the reference)
// ---------------------------------------------------------------------------
#define BATCH 2
#define SEQ   2048
#define DXDIM 512
#define NHEAD 8
#define DHEAD 64
#define MROWS (BATCH * SEQ)          // 4096 rows in all projection GEMMs

typedef __attribute__((ext_vector_type(16))) __bf16 v16bf;
typedef __attribute__((ext_vector_type(8)))  __bf16 v8bf;
typedef __attribute__((ext_vector_type(8)))  float  v8f;

__device__ __forceinline__ v16bf ld_frag16(const __bf16* p0, const __bf16* p1) {
  v8bf lo = *reinterpret_cast<const v8bf*>(p0);
  v8bf hi = *reinterpret_cast<const v8bf*>(p1);
  v16bf r;
#pragma unroll
  for (int i = 0; i < 8; ++i) { r[i] = lo[i]; r[i + 8] = hi[i]; }
  return r;
}

__device__ __forceinline__ v8f wmma_bf16(v16bf a, v16bf b, v8f c) {
  // D = A(16x32 bf16) * B(32x16 bf16) + C(16x16 f32)
  return __builtin_amdgcn_wmma_f32_16x16x32_bf16(
      /*neg_a=*/false, a, /*neg_b=*/false, b,
      /*c_mod=*/(short)0, c, /*reuse_a=*/false, /*reuse_b=*/false);
}

// ---------------------------------------------------------------------------
// Elementwise f32 -> bf16 convert
// ---------------------------------------------------------------------------
__global__ void cvt_f32_bf16(const float* __restrict__ src,
                             __bf16* __restrict__ dst, int n) {
  int i = blockIdx.x * blockDim.x + threadIdx.x;
  if (i < n) dst[i] = (__bf16)src[i];
}

// Transpose + convert a 512x512 f32 weight: Wt[n*512 + k] = W[k*512 + n]
__global__ void transpose_cvt_512(const float* __restrict__ W,
                                  __bf16* __restrict__ Wt) {
  int idx = blockIdx.x * blockDim.x + threadIdx.x;   // 512*512 threads
  int n = idx & 511;
  int k = idx >> 9;
  Wt[(size_t)n * DXDIM + k] = (__bf16)W[(size_t)k * DXDIM + n];
}

// ---------------------------------------------------------------------------
// Generic WMMA GEMM: C[M=4096, N=512] = A[4096,512](bf16) * W[512,512] + bias
//   Wt is the transposed bf16 weight (N-major, K contiguous).
//   Each wave computes a 16x64 tile: one A fragment reused across 4
//   independent accumulator chains (no WMMA->WMMA RAW serialization).
//   mode 0: out bf16 head-major  [B, H, T, DH], value scaled by `scale`
//   mode 1: out f32  flat        [M, 512]
//   mode 2: out bf16 head-major TRANSPOSED [B, H, DH, T]
// ---------------------------------------------------------------------------
__global__ void gemm_wmma_bf16(const __bf16* __restrict__ A,
                               const __bf16* __restrict__ Wt,
                               const float*  __restrict__ bias,
                               float scale, int mode,
                               float* __restrict__ outF,
                               __bf16* __restrict__ outB) {
  const int lane = threadIdx.x & 31;
  const int wave = threadIdx.x >> 5;
  const int job  = blockIdx.x * 8 + wave;      // 256 m-tiles * 8 n-groups = 2048
  const int ng = job & 7;                      // 512/64 = 8 n-groups
  const int mt = job >> 3;                     // 4096/16 = 256 m-tiles
  const int m0 = mt * 16, n0 = ng * 64;
  const int lm = lane & 15;
  const int half = lane >> 4;                  // 0 or 1

  // A fragment: row (m0+lm), K chunk koff16..+15
  const __bf16* ap = A + (size_t)(m0 + lm) * DXDIM + half * 16;
  // B fragments: Wt row = output column (n0 + c*16 + lm), K chunk koff16..+15
  const __bf16* bp0 = Wt + (size_t)(n0 + 0  + lm) * DXDIM + half * 16;
  const __bf16* bp1 = Wt + (size_t)(n0 + 16 + lm) * DXDIM + half * 16;
  const __bf16* bp2 = Wt + (size_t)(n0 + 32 + lm) * DXDIM + half * 16;
  const __bf16* bp3 = Wt + (size_t)(n0 + 48 + lm) * DXDIM + half * 16;

  // Pull the A row into the near caches while WMMAs run.
  __builtin_prefetch(ap, 0, 3);
  __builtin_prefetch(ap + 256, 0, 3);

  v8f acc0 = {}, acc1 = {}, acc2 = {}, acc3 = {};
#pragma unroll 2
  for (int k0 = 0; k0 < DXDIM; k0 += 32) {
    v16bf af = ld_frag16(ap + k0, ap + k0 + 8);
    v16bf b0 = ld_frag16(bp0 + k0, bp0 + k0 + 8);
    v16bf b1 = ld_frag16(bp1 + k0, bp1 + k0 + 8);
    v16bf b2 = ld_frag16(bp2 + k0, bp2 + k0 + 8);
    v16bf b3 = ld_frag16(bp3 + k0, bp3 + k0 + 8);
    acc0 = wmma_bf16(af, b0, acc0);
    acc1 = wmma_bf16(af, b1, acc1);
    acc2 = wmma_bf16(af, b2, acc2);
    acc3 = wmma_bf16(af, b3, acc3);
  }

  v8f accs[4] = {acc0, acc1, acc2, acc3};
#pragma unroll
  for (int c = 0; c < 4; ++c) {
    const int N = n0 + c * 16 + lm;
    const float bval = bias[N];
#pragma unroll
    for (int r = 0; r < 8; ++r) {
      const int M = m0 + r + half * 8;         // C layout: VGPR r -> M=r / r+8
      float v = accs[c][r] + bval;
      if (mode == 1) {
        outF[(size_t)M * DXDIM + N] = v;
      } else {
        const int bb = M >> 11, t = M & (SEQ - 1);     // T = 2048
        const int hh = N >> 6,  d = N & (DHEAD - 1);   // DH = 64
        if (mode == 0) {
          outB[(((size_t)(bb * NHEAD + hh) * SEQ) + t) * DHEAD + d] =
              (__bf16)(v * scale);
        } else { // mode 2: [B,H,DH,T]
          outB[(((size_t)(bb * NHEAD + hh) * DHEAD) + d) * SEQ + t] = (__bf16)v;
        }
      }
    }
  }
}

// ---------------------------------------------------------------------------
// Attention: one wave (32 threads) per (b, h, 16-row query tile).
//   S stripe 16x2048 f32 lives in dynamic LDS (128 KB of the 320 KB WGP LDS).
//   Pass 1: S = (Q/sqrt(64)) * K^T via WMMA   (Q pre-scaled at projection)
//   Pass 2: row max / row sumexp (lane pairs + shfl_xor 16)
//   Pass 3: v_bar = softmax(S) * V via WMMA, gate with R, emit bf16 new_v
// ---------------------------------------------------------------------------
#define SMEM_S_BYTES (16 * SEQ * 4)
#define SMEM_BYTES   (SMEM_S_BYTES + 32 * 4)

__global__ void attn_wmma(const __bf16* __restrict__ qhm,
                          const __bf16* __restrict__ khm,
                          const __bf16* __restrict__ vhmT,
                          const float*  __restrict__ rflat,
                          __bf16* __restrict__ nv) {
  extern __shared__ char smem[];
  float* S      = reinterpret_cast<float*>(smem);
  float* rowmax = reinterpret_cast<float*>(smem + SMEM_S_BYTES);
  float* rowinv = rowmax + 16;

  const int lane = threadIdx.x;
  const int blk  = blockIdx.x;                 // B*H*(T/16) = 2048 blocks
  const int it = blk & 127;                    // T/16 = 128 row tiles
  const int h  = (blk >> 7) & (NHEAD - 1);
  const int b  = blk >> 10;
  const int i0 = it * 16;

  const int lm = lane & 15;
  const int half = lane >> 4;
  const int koff8  = half * 8;
  const int koff16 = half * 16;

  const __bf16* qbase = qhm  + ((size_t)(b * NHEAD + h) * SEQ) * DHEAD;
  const __bf16* kbase = khm  + ((size_t)(b * NHEAD + h) * SEQ) * DHEAD;
  const __bf16* vbase = vhmT + ((size_t)(b * NHEAD + h) * DHEAD) * SEQ;

  // Q fragments (rows i0..i0+15, K = d), reused for every key tile
  v16bf qf[2];
#pragma unroll
  for (int kk = 0; kk < 2; ++kk) {
    const __bf16* p = qbase + (size_t)(i0 + lm) * DHEAD + kk * 32 + koff8;
    qf[kk] = ld_frag16(p, p + 16);
  }

  // ---- Pass 1: scores (unroll 2 j-tiles -> 2 independent WMMA chains) ----
#pragma unroll 2
  for (int jt = 0; jt < SEQ / 16; ++jt) {
    v8f acc = {};
#pragma unroll
    for (int kk = 0; kk < 2; ++kk) {
      // B frag: column j = jt*16+lm, contiguous d (head dim)
      const __bf16* p = kbase + (size_t)(jt * 16 + lm) * DHEAD + kk * 32 + koff8;
      v16bf bf = ld_frag16(p, p + 16);
      acc = wmma_bf16(qf[kk], bf, acc);
    }
#pragma unroll
    for (int r = 0; r < 8; ++r)
      S[(r + half * 8) * SEQ + jt * 16 + lm] = acc[r];
  }
  __syncthreads();

  // ---- Pass 2: row stats (lane handles row lm, half the columns) ----
  {
    const int base = lm * SEQ + half * (SEQ / 2);
    float mx = -3.4e38f;
    for (int c = 0; c < SEQ / 2; c += 4) {
      float4 v = *reinterpret_cast<float4*>(&S[base + c]);
      mx = fmaxf(mx, fmaxf(fmaxf(v.x, v.y), fmaxf(v.z, v.w)));
    }
    mx = fmaxf(mx, __shfl_xor(mx, 16));
    float sum = 0.f;
    for (int c = 0; c < SEQ / 2; c += 4) {
      float4 v = *reinterpret_cast<float4*>(&S[base + c]);
      sum += __expf(v.x - mx) + __expf(v.y - mx) +
             __expf(v.z - mx) + __expf(v.w - mx);
    }
    sum += __shfl_xor(sum, 16);
    if (half == 0) { rowmax[lm] = mx; rowinv[lm] = 1.0f / sum; }
  }
  __syncthreads();

  // ---- Pass 3: v_bar = P * V (4 independent accumulator chains) ----
  v8f acc[4] = {v8f{}, v8f{}, v8f{}, v8f{}};
  const float rm = rowmax[lm];
  for (int jb = 0; jb < SEQ; jb += 32) {
    // A frag: row lm of P, K chunks {jb+koff8..+7} and {jb+16+koff8..+7}
    const int base0 = lm * SEQ + jb + koff8;
    float4 s0 = *reinterpret_cast<float4*>(&S[base0]);
    float4 s1 = *reinterpret_cast<float4*>(&S[base0 + 4]);
    float4 s2 = *reinterpret_cast<float4*>(&S[base0 + 16]);
    float4 s3 = *reinterpret_cast<float4*>(&S[base0 + 20]);
    v16bf af;
    af[0]  = (__bf16)__expf(s0.x - rm); af[1]  = (__bf16)__expf(s0.y - rm);
    af[2]  = (__bf16)__expf(s0.z - rm); af[3]  = (__bf16)__expf(s0.w - rm);
    af[4]  = (__bf16)__expf(s1.x - rm); af[5]  = (__bf16)__expf(s1.y - rm);
    af[6]  = (__bf16)__expf(s1.z - rm); af[7]  = (__bf16)__expf(s1.w - rm);
    af[8]  = (__bf16)__expf(s2.x - rm); af[9]  = (__bf16)__expf(s2.y - rm);
    af[10] = (__bf16)__expf(s2.z - rm); af[11] = (__bf16)__expf(s2.w - rm);
    af[12] = (__bf16)__expf(s3.x - rm); af[13] = (__bf16)__expf(s3.y - rm);
    af[14] = (__bf16)__expf(s3.z - rm); af[15] = (__bf16)__expf(s3.w - rm);
#pragma unroll
    for (int c = 0; c < 4; ++c) {
      // B frag: column d = c*16+lm of V, contiguous j (from transposed V)
      const __bf16* p = vbase + (size_t)(c * 16 + lm) * SEQ + jb + koff8;
      v16bf bf = ld_frag16(p, p + 16);
      acc[c] = wmma_bf16(af, bf, acc[c]);
    }
  }

  // ---- Epilogue: normalize, gate with R, write new_v (bf16, flat) ----
  float inv[8];
#pragma unroll
  for (int r = 0; r < 8; ++r) inv[r] = rowinv[r + half * 8];
#pragma unroll
  for (int c = 0; c < 4; ++c) {
#pragma unroll
    for (int r = 0; r < 8; ++r) {
      const int i = i0 + r + half * 8;
      const int dfull = h * DHEAD + c * 16 + lm;
      const size_t off = ((size_t)(b * SEQ + i)) * DXDIM + dfull;
      const float val = acc[c][r] * inv[r] * rflat[off];
      nv[off] = (__bf16)val;
    }
  }
}

// ---------------------------------------------------------------------------
// Host launcher
// ---------------------------------------------------------------------------
extern "C" void kernel_launch(void* const* d_in, const int* in_sizes, int n_in,
                              void* d_out, int out_size, void* d_ws, size_t ws_size,
                              hipStream_t stream) {
  const float* q  = (const float*)d_in[0];
  const float* k  = (const float*)d_in[1];
  const float* v  = (const float*)d_in[2];
  const float* Wq = (const float*)d_in[3];
  const float* bq = (const float*)d_in[4];
  const float* Wk = (const float*)d_in[5];
  const float* bk = (const float*)d_in[6];
  const float* Wv = (const float*)d_in[7];
  const float* bv = (const float*)d_in[8];
  const float* Wr = (const float*)d_in[9];
  const float* br = (const float*)d_in[10];
  const float* Wo = (const float*)d_in[11];
  const float* bo = (const float*)d_in[12];
  float* out = (float*)d_out;

  // Workspace carve-out (256-byte aligned)
  char* ws = (char*)d_ws;
  size_t off = 0;
  auto alloc = [&](size_t bytes) -> char* {
    char* p = ws + off;
    off += (bytes + 255) & ~(size_t)255;
    return p;
  };
  const size_t NTOK = (size_t)MROWS * DXDIM;      // 4096*512
  __bf16* qbf  = (__bf16*)alloc(NTOK * 2);
  __bf16* kbf  = (__bf16*)alloc(NTOK * 2);
  __bf16* vbf  = (__bf16*)alloc(NTOK * 2);
  __bf16* wqT  = (__bf16*)alloc((size_t)DXDIM * DXDIM * 2);
  __bf16* wkT  = (__bf16*)alloc((size_t)DXDIM * DXDIM * 2);
  __bf16* wvT  = (__bf16*)alloc((size_t)DXDIM * DXDIM * 2);
  __bf16* wrT  = (__bf16*)alloc((size_t)DXDIM * DXDIM * 2);
  __bf16* woT  = (__bf16*)alloc((size_t)DXDIM * DXDIM * 2);
  __bf16* qhm  = (__bf16*)alloc(NTOK * 2);        // [B,H,T,DH] bf16 (pre-scaled)
  __bf16* khm  = (__bf16*)alloc(NTOK * 2);        // [B,H,T,DH] bf16
  __bf16* vhmT = (__bf16*)alloc(NTOK * 2);        // [B,H,DH,T] bf16
  float*  rfl  = (float*) alloc(NTOK * 4);        // [B,T,512] f32 gate
  __bf16* nv   = (__bf16*)alloc(NTOK * 2);        // [B,T,512] bf16 gated v_bar
  (void)ws_size; (void)in_sizes; (void)n_in; (void)out_size;

  // 1) bf16 conversions
  const int ncv = (int)NTOK;
  cvt_f32_bf16<<<(ncv + 255) / 256, 256, 0, stream>>>(q, qbf, ncv);
  cvt_f32_bf16<<<(ncv + 255) / 256, 256, 0, stream>>>(k, kbf, ncv);
  cvt_f32_bf16<<<(ncv + 255) / 256, 256, 0, stream>>>(v, vbf, ncv);
  const int nw = DXDIM * DXDIM;
  transpose_cvt_512<<<nw / 256, 256, 0, stream>>>(Wq, wqT);
  transpose_cvt_512<<<nw / 256, 256, 0, stream>>>(Wk, wkT);
  transpose_cvt_512<<<nw / 256, 256, 0, stream>>>(Wv, wvT);
  transpose_cvt_512<<<nw / 256, 256, 0, stream>>>(Wr, wrT);
  transpose_cvt_512<<<nw / 256, 256, 0, stream>>>(Wo, woT);

  // 2) projections: 2048 wave-jobs (16x64 tiles) / 8 waves = 256 blocks
  const int gblks = (MROWS / 16) * (DXDIM / 64) / 8;
  gemm_wmma_bf16<<<gblks, 256, 0, stream>>>(qbf, wqT, bq, 0.125f, 0, nullptr, qhm);
  gemm_wmma_bf16<<<gblks, 256, 0, stream>>>(kbf, wkT, bk, 1.0f,   0, nullptr, khm);
  gemm_wmma_bf16<<<gblks, 256, 0, stream>>>(vbf, wvT, bv, 1.0f,   2, nullptr, vhmT);
  gemm_wmma_bf16<<<gblks, 256, 0, stream>>>(qbf, wrT, br, 1.0f,   1, rfl, nullptr);

  // 3) attention: B*H*(T/16) = 2048 single-wave blocks, 128 KB LDS each
  attn_wmma<<<BATCH * NHEAD * (SEQ / 16), 32, SMEM_BYTES, stream>>>(
      qhm, khm, vhmT, rfl, nv);

  // 4) output projection -> f32 d_out
  gemm_wmma_bf16<<<gblks, 256, 0, stream>>>(nv, woT, bo, 1.0f, 1, out, nullptr);
}
